// SpatialAttention_52707838656852
// MI455X (gfx1250) — matrix operand order repaired and checked by reading, not compiled, not run on hardware
//
#include <hip/hip_runtime.h>
#include <hip/hip_bf16.h>

typedef __attribute__((ext_vector_type(16))) _Float16 v16h;
typedef __attribute__((ext_vector_type(8)))  _Float16 v8h;
typedef __attribute__((ext_vector_type(4)))  _Float16 v4h;
typedef __attribute__((ext_vector_type(8)))  float    v8f;

#define C_DIM 512
#define HW    1024
#define BATCH 16
#define M_TOK (BATCH * HW) // 16384

// ---------------------------------------------------------------------------
// WMMA fragment helpers (layouts per CDNA5 ISA 7.12.2, wave32)
// ---------------------------------------------------------------------------

// A fragment: 16x32 f16, A row-major with leading dim lda (in halfs).
// lane L (0..15): row M=L, K = {0..7, 16..23}; lane 16..31: row M=L-16, K = {8..15, 24..31}
__device__ __forceinline__ v16h load_a_frag(const _Float16* __restrict__ a, int lda) {
    int lane = threadIdx.x & 31;
    const _Float16* p = a + (size_t)(lane & 15) * lda + ((lane >> 4) << 3);
    v8h lo = *(const v8h*)p;        // K base .. +7
    v8h hi = *(const v8h*)(p + 16); // K base+16 .. +23
    v16h f;
#pragma unroll
    for (int i = 0; i < 8; ++i) { f[i] = lo[i]; f[i + 8] = hi[i]; }
    return f;
}

// B fragment: 32x16 f16 of B^T where memory holds B as [N,K] row-major (ldb in halfs).
// lane L (0..15): col N=L, K = {0..7, 16..23}; lane 16..31: col N=L-16, K = {8..15, 24..31}
__device__ __forceinline__ v16h load_b_frag(const _Float16* __restrict__ b, int ldb) {
    int lane = threadIdx.x & 31;
    const _Float16* p = b + (size_t)(lane & 15) * ldb + ((lane >> 4) << 3);
    v8h lo = *(const v8h*)p;
    v8h hi = *(const v8h*)(p + 16);
    v16h f;
#pragma unroll
    for (int i = 0; i < 8; ++i) { f[i] = lo[i]; f[i + 8] = hi[i]; }
    return f;
}

__device__ __forceinline__ v8f wmma_f16(const v16h& a, const v16h& b, const v8f& c) {
    return __builtin_amdgcn_wmma_f32_16x16x32_f16(
        /*neg_a=*/false, a, /*neg_b=*/false, b,
        /*c_mod=*/(short)0, c, /*reuse_a=*/false, /*reuse_b=*/false);
}

// One wave computes a 32x64 f32 tile of A[M,K] x B[N,K]^T. KTILES = K/32.
template <int KTILES>
__device__ __forceinline__ void gemm_32x64(const _Float16* __restrict__ A, int lda,
                                           const _Float16* __restrict__ B, int ldb,
                                           v8f acc[2][4]) {
#pragma unroll 1
    for (int kk = 0; kk < KTILES; ++kk) {
        const _Float16* ap = A + kk * 32;
        const _Float16* bp = B + kk * 32;
        v16h a0 = load_a_frag(ap, lda);
        v16h a1 = load_a_frag(ap + (size_t)16 * lda, lda);
#pragma unroll
        for (int j = 0; j < 4; ++j) {
            v16h bf = load_b_frag(bp + (size_t)(j * 16) * ldb, ldb);
            acc[0][j] = wmma_f16(a0, bf, acc[0][j]);
            acc[1][j] = wmma_f16(a1, bf, acc[1][j]);
        }
    }
}

// C fragment element mapping helper: lane -> (n = lane&15, m = 8*(lane>>4) + v)

// ---------------------------------------------------------------------------
// Kernel 0: weight transpose + f32 -> f16   (Wqkv[c,n,3] -> WT[e][n][k], Wout[c,d] -> WoT[d][c])
// ---------------------------------------------------------------------------
__global__ void prep_weights_kernel(const float* __restrict__ Wqkv, const float* __restrict__ Wout,
                                    _Float16* __restrict__ WT, _Float16* __restrict__ WoT) {
    int tid = blockIdx.x * 256 + threadIdx.x;
    if (tid < 3 * C_DIM * C_DIM) {
        int e = tid / (C_DIM * C_DIM);
        int rem = tid - e * (C_DIM * C_DIM);
        int n = rem >> 9, k = rem & 511;
        WT[(size_t)e * C_DIM * C_DIM + (size_t)n * C_DIM + k] =
            (_Float16)Wqkv[((size_t)k * C_DIM + n) * 3 + e];
    } else {
        int rem = tid - 3 * C_DIM * C_DIM;
        int n = rem >> 9, k = rem & 511;
        WoT[(size_t)n * C_DIM + k] = (_Float16)Wout[(size_t)k * C_DIM + n];
    }
}

// ---------------------------------------------------------------------------
// Kernel 1: GroupNorm -> f16 normalized activations. One block per (batch, group).
// ---------------------------------------------------------------------------
__global__ void groupnorm_kernel(const float* __restrict__ x, const float* __restrict__ gamma,
                                 const float* __restrict__ beta, _Float16* __restrict__ xn) {
    __shared__ float s1[256], s2[256];
    int bid = blockIdx.x;
    int b = bid >> 5, g = bid & 31;
    int tid = threadIdx.x;
    const float* xb = x + (size_t)b * HW * C_DIM + g * 16;
    float sum = 0.f, sumsq = 0.f;
    for (int li = tid; li < 16384; li += 256) {
        int s = li >> 4, c = li & 15;
        float v = xb[(size_t)s * C_DIM + c];
        sum += v; sumsq += v * v;
    }
    s1[tid] = sum; s2[tid] = sumsq;
    __syncthreads();
    for (int off = 128; off > 0; off >>= 1) {
        if (tid < off) { s1[tid] += s1[tid + off]; s2[tid] += s2[tid + off]; }
        __syncthreads();
    }
    float mean = s1[0] * (1.0f / 16384.0f);
    float var  = s2[0] * (1.0f / 16384.0f) - mean * mean;
    float inv  = rsqrtf(var + 1e-3f);
    _Float16* xnb = xn + (size_t)b * HW * C_DIM + g * 16;
    for (int li = tid; li < 16384; li += 256) {
        int s = li >> 4, c = li & 15;
        float v = xb[(size_t)s * C_DIM + c];
        float r = (v - mean) * inv * gamma[g * 16 + c] + beta[g * 16 + c];
        xnb[(size_t)s * C_DIM + c] = (_Float16)r;
    }
}

// ---------------------------------------------------------------------------
// Kernel 2: QKV GEMM. Xn[16384,512] x WT[e][512n][512k]^T. Q scaled by C^-0.5,
// Q/K stored row-major f16, V stored transposed as Vt[b][d][key] f16.
// ---------------------------------------------------------------------------
__global__ void qkv_gemm_kernel(const _Float16* __restrict__ Xn, const _Float16* __restrict__ WT,
                                const float* __restrict__ bqkv, _Float16* __restrict__ Q,
                                _Float16* __restrict__ Kf, _Float16* __restrict__ Vt) {
    int wave = blockIdx.x * 8 + (threadIdx.x >> 5);
    if (wave >= 512 * 24) return;
    int mt = wave / 24, nt = wave - mt * 24;
    int m0 = mt * 32;
    int col0 = nt * 64;
    int e = col0 >> 9;      // 0=q, 1=k, 2=v (each 64-wide tile is within one e)
    int n0 = col0 & 511;
    v8f acc[2][4] = {};
    gemm_32x64<16>(Xn + (size_t)m0 * C_DIM, C_DIM,
                   WT + (size_t)e * C_DIM * C_DIM + (size_t)n0 * C_DIM, C_DIM, acc);
    int lane = threadIdx.x & 31;
    int cn = lane & 15, cm = (lane >> 4) * 8;
    const float scale = 0.044194173824159216f; // 512^-0.5 folded into Q
    _Float16* dst = (e == 0) ? Q : Kf;
#pragma unroll
    for (int i = 0; i < 2; ++i)
#pragma unroll
        for (int j = 0; j < 4; ++j)
#pragma unroll
            for (int v = 0; v < 8; ++v) {
                int m = m0 + i * 16 + cm + v;
                int n = n0 + j * 16 + cn;
                float val = acc[i][j][v] + bqkv[n * 3 + e];
                if (e == 0) val *= scale;
                if (e < 2)
                    dst[(size_t)m * C_DIM + n] = (_Float16)val;
                else
                    Vt[((size_t)(m >> 10) * C_DIM + n) * HW + (m & 1023)] = (_Float16)val;
            }
}

// ---------------------------------------------------------------------------
// Kernel 3: S = Q K^T (scale pre-folded into Q). Per batch [1024,512]x[1024,512]^T.
// S f32 lives in workspace (64 MB total -> resident in 192 MB L2).
// ---------------------------------------------------------------------------
__global__ void scores_kernel(const _Float16* __restrict__ Q, const _Float16* __restrict__ Kf,
                              float* __restrict__ S) {
    int wave = blockIdx.x * 8 + (threadIdx.x >> 5);
    if (wave >= BATCH * 512) return;
    int b = wave >> 9, r = wave & 511;
    int mt = r >> 4, nt = r & 15;
    int m0 = mt * 32, n0 = nt * 64;
    v8f acc[2][4] = {};
    gemm_32x64<16>(Q + ((size_t)b * HW + m0) * C_DIM, C_DIM,
                   Kf + ((size_t)b * HW + n0) * C_DIM, C_DIM, acc);
    int lane = threadIdx.x & 31;
    int cn = lane & 15, cm = (lane >> 4) * 8;
    float* Sb = S + (size_t)b * HW * HW;
#pragma unroll
    for (int i = 0; i < 2; ++i)
#pragma unroll
        for (int j = 0; j < 4; ++j)
#pragma unroll
            for (int v = 0; v < 8; ++v)
                Sb[(size_t)(m0 + i * 16 + cm + v) * HW + (n0 + j * 16 + cn)] = acc[i][j][v];
}

// ---------------------------------------------------------------------------
// Kernel 4: row softmax over 1024 keys; writes P f16 row-major (A-fragment friendly).
// One block per row, 4 elements per thread.
// ---------------------------------------------------------------------------
__global__ void softmax_kernel(const float* __restrict__ S, _Float16* __restrict__ P) {
    __shared__ float red[256];
    int row = blockIdx.x, tid = threadIdx.x;
    const float4* sr = (const float4*)(S + (size_t)row * HW);
    float4 v = sr[tid];
    float m = fmaxf(fmaxf(v.x, v.y), fmaxf(v.z, v.w));
    red[tid] = m; __syncthreads();
    for (int off = 128; off > 0; off >>= 1) {
        if (tid < off) red[tid] = fmaxf(red[tid], red[tid + off]);
        __syncthreads();
    }
    float rowmax = red[0];
    __syncthreads();
    float e0 = __expf(v.x - rowmax), e1 = __expf(v.y - rowmax);
    float e2 = __expf(v.z - rowmax), e3 = __expf(v.w - rowmax);
    red[tid] = e0 + e1 + e2 + e3; __syncthreads();
    for (int off = 128; off > 0; off >>= 1) {
        if (tid < off) red[tid] += red[tid + off];
        __syncthreads();
    }
    float inv = 1.0f / red[0];
    v4h o;
    o[0] = (_Float16)(e0 * inv); o[1] = (_Float16)(e1 * inv);
    o[2] = (_Float16)(e2 * inv); o[3] = (_Float16)(e3 * inv);
    *((v4h*)(P + (size_t)row * HW) + tid) = o;
}

// ---------------------------------------------------------------------------
// Kernel 5: O = P V. Per batch [1024,1024] x Vt[512,1024]^T -> O f16 [1024,512].
// ---------------------------------------------------------------------------
__global__ void av_kernel(const _Float16* __restrict__ P, const _Float16* __restrict__ Vt,
                          _Float16* __restrict__ O) {
    int wave = blockIdx.x * 8 + (threadIdx.x >> 5);
    if (wave >= BATCH * 256) return;
    int b = wave >> 8, r = wave & 255;
    int mt = r >> 3, nt = r & 7;
    int m0 = mt * 32, n0 = nt * 64;
    v8f acc[2][4] = {};
    gemm_32x64<32>(P + ((size_t)b * HW + m0) * HW, HW,
                   Vt + ((size_t)b * C_DIM + n0) * HW, HW, acc);
    int lane = threadIdx.x & 31;
    int cn = lane & 15, cm = (lane >> 4) * 8;
    _Float16* Ob = O + (size_t)b * HW * C_DIM;
#pragma unroll
    for (int i = 0; i < 2; ++i)
#pragma unroll
        for (int j = 0; j < 4; ++j)
#pragma unroll
            for (int v = 0; v < 8; ++v)
                Ob[(size_t)(m0 + i * 16 + cm + v) * C_DIM + (n0 + j * 16 + cn)] =
                    (_Float16)acc[i][j][v];
}

// ---------------------------------------------------------------------------
// Kernel 6: out = inputs + O @ Wout + b_out (f32 store)
// ---------------------------------------------------------------------------
__global__ void out_proj_kernel(const _Float16* __restrict__ O, const _Float16* __restrict__ WoT,
                                const float* __restrict__ bout, const float* __restrict__ inp,
                                float* __restrict__ out) {
    int wave = blockIdx.x * 8 + (threadIdx.x >> 5);
    if (wave >= 512 * 8) return;
    int mt = wave >> 3, nt = wave & 7;
    int m0 = mt * 32, n0 = nt * 64;
    v8f acc[2][4] = {};
    gemm_32x64<16>(O + (size_t)m0 * C_DIM, C_DIM, WoT + (size_t)n0 * C_DIM, C_DIM, acc);
    int lane = threadIdx.x & 31;
    int cn = lane & 15, cm = (lane >> 4) * 8;
#pragma unroll
    for (int i = 0; i < 2; ++i)
#pragma unroll
        for (int j = 0; j < 4; ++j)
#pragma unroll
            for (int v = 0; v < 8; ++v) {
                size_t m = (size_t)(m0 + i * 16 + cm + v);
                int n = n0 + j * 16 + cn;
                out[m * C_DIM + n] = inp[m * C_DIM + n] + acc[i][j][v] + bout[n];
            }
}

// ---------------------------------------------------------------------------
// Launch: workspace layout (130 MB, with aliasing of dead buffers):
//   [0,16MB)    Xn f16            -> reused as O after QKV GEMM consumed it
//   [16,32MB)   Q f16             -> reused (with K region) as P after scores
//   [32,48MB)   K f16
//   [48,64MB)   Vt f16 (V transposed [b][d][key])
//   [64,128MB)  S f32
//   [128MB,..)  WT f16 (3x512x512) + WoT f16 (512x512)
// ---------------------------------------------------------------------------
extern "C" void kernel_launch(void* const* d_in, const int* in_sizes, int n_in,
                              void* d_out, int out_size, void* d_ws, size_t ws_size,
                              hipStream_t stream) {
    const float* inp   = (const float*)d_in[0];
    const float* gamma = (const float*)d_in[1];
    const float* beta  = (const float*)d_in[2];
    const float* Wqkv  = (const float*)d_in[3];
    const float* bqkv  = (const float*)d_in[4];
    const float* Wout  = (const float*)d_in[5];
    const float* bout  = (const float*)d_in[6];
    float* out = (float*)d_out;

    char* ws = (char*)d_ws;
    const size_t MB = 1024 * 1024;
    _Float16* Xn  = (_Float16*)(ws + 0);
    _Float16* Q   = (_Float16*)(ws + 16 * MB);
    _Float16* Kf  = (_Float16*)(ws + 32 * MB);
    _Float16* Vt  = (_Float16*)(ws + 48 * MB);
    float*    S   = (float*)   (ws + 64 * MB);
    _Float16* WT  = (_Float16*)(ws + 128 * MB);
    _Float16* WoT = (_Float16*)(ws + 128 * MB + (size_t)3 * C_DIM * C_DIM * 2);
    _Float16* P   = Q;   // aliases Q|K region (dead after scores)
    _Float16* O   = Xn;  // aliases Xn region (dead after QKV)

    prep_weights_kernel<<<4096, 256, 0, stream>>>(Wqkv, Wout, WT, WoT);
    groupnorm_kernel<<<BATCH * 32, 256, 0, stream>>>(inp, gamma, beta, Xn);
    qkv_gemm_kernel<<<1536, 256, 0, stream>>>(Xn, WT, bqkv, Q, Kf, Vt);
    scores_kernel<<<1024, 256, 0, stream>>>(Q, Kf, S);
    softmax_kernel<<<BATCH * HW, 256, 0, stream>>>(S, P);
    av_kernel<<<512, 256, 0, stream>>>(P, Vt, O);
    out_proj_kernel<<<512, 256, 0, stream>>>(O, WoT, bout, inp, out);
}
